// TransformerEncoder_53532472377425
// MI455X (gfx1250) — matrix operand order repaired and checked by reading, not compiled, not run on hardware
//
#include <hip/hip_runtime.h>

// ---------------- problem constants ----------------
#define BB   512
#define SS   47
#define LLQ  48      // sequence length incl. CLS
#define DD   32      // embed dim
#define HHN  4       // heads
#define DH   8       // head dim
#define FFD  2048
#define HIDD 256

typedef __attribute__((ext_vector_type(16))) _Float16 v16h;
typedef __attribute__((ext_vector_type(8)))  float    v8f;

// Fixed positional buffer from the reference (47 x 3)
__constant__ float c_pos[SS * 3] = {
  0,3,0.5f, 0,3,1.5f, 0,3,2.5f, 0,3,3.5f, 0,3,4.5f, 0,3,5.5f,
  1,0,0, 1,6,6, 2,0,6, 2,6,0, 3,0,0, 3,6,6, 4,0,6, 4,6,0,
  5,0,0, 5,6,6, 6,0,6, 6,6,0, 7,0,0, 7,6,6, 8,0,6, 8,6,0,
  9,0,0, 9,6,6, 10,0,6, 10,6,0, 11,0,0, 11,6,6, 12,0,6, 12,6,0,
  13,0,0, 13,6,6, 14,0,6, 14,6,0, 15,0,0, 15,6,6, 16,0,6, 16,6,0,
  17,0.5f,0.5f, 17,3,0.5f, 17,5.5f,0.5f,
  17,0.5f,3,    17,3,3,    17,5.5f,3,
  17,0.5f,5.5f, 17,3,5.5f, 17,5.5f,5.5f
};

// ---------------- WMMA helpers (gfx1250, wave32) ----------------
__device__ __forceinline__ v8f wmma32(v16h a, v16h b, v8f c) {
  // D = A(16x32 f16) * B(32x16 f16) + C(16x16 f32)
  return __builtin_amdgcn_wmma_f32_16x16x32_f16(
      /*neg_a=*/false, a, /*neg_b=*/false, b,
      /*c_mod=*/(short)0, c, /*reuse_a=*/false, /*reuse_b=*/false);
}

// A tile loader: A[m][k] from row-major f32 buffer (ld = row stride).
// Contiguous-K A-layout used here (same as B loaders): lane L holds row M=L&15; elements 0..7 -> K = (L<16?0:16)+e,
// elements 8..15 -> K = (L<16?8:24)+(e-8).  kvalid counts valid K from k0.
__device__ __forceinline__ v16h load_a16(const float* base, int ld, int m0,
                                         int k0, int lane, int kvalid) {
  const float* row = base + (m0 + (lane & 15)) * ld + k0;
  const int klo = (lane & 16) ? 16 : 0;
  const int khi = (lane & 16) ? 24 : 8;
  v16h a;
#pragma unroll
  for (int e = 0; e < 8; ++e) {
    int k1 = klo + e, k2 = khi + e;
    a[e]     = (_Float16)(k1 < kvalid ? row[k1] : 0.0f);
    a[e + 8] = (_Float16)(k2 < kvalid ? row[k2] : 0.0f);
  }
  return a;
}

// B tile loader from "weight style" storage: B[k][n] = W[(n0+n)*ldw + k0+k].
// Contiguous-K B-layout (same as load_a16): lane L holds col N=L&15; lanes 0-15 K=0..15, lanes 16-31 K=16..31.
__device__ __forceinline__ v16h load_bw16(const float* W, int ldw, int n0,
                                          int k0, int lane, int kvalid, int nvalid) {
  const int n  = lane & 15;
  const int kb = (lane & 16) ? 16 : 0;
  const bool nok = n < nvalid;
  const float* row = W + (n0 + n) * ldw + k0;
  v16h b;
#pragma unroll
  for (int e = 0; e < 16; ++e) {
    int k = kb + e;
    b[e] = (_Float16)((nok && k < kvalid) ? row[k] : 0.0f);
  }
  return b;
}

// B tile loader from k-major storage: B[k][n] = M[(k0+k)*ldm + n0+n].
__device__ __forceinline__ v16h load_bm16(const float* M, int ldm, int k0,
                                          int n0, int lane, int kvalid, int nvalid) {
  const int n  = lane & 15;
  const int kb = (lane & 16) ? 16 : 0;
  const bool nok = n < nvalid;
  v16h b;
#pragma unroll
  for (int e = 0; e < 16; ++e) {
    int k = kb + e;
    b[e] = (_Float16)((nok && k < kvalid) ? M[(k0 + k) * ldm + n0 + n] : 0.0f);
  }
  return b;
}

// C/D store: VGPR r -> row m0 + r + (lane>=16 ? 8 : 0), col n0 + (lane&15).
__device__ __forceinline__ void store_c(float* dst, int ld, int m0, int n0,
                                        int lane, v8f c, int nvalid, float scale,
                                        const float* bias /*per-col in tile or null*/,
                                        bool relu) {
  const int n = lane & 15;
  if (n >= nvalid) return;
  const float bv = bias ? bias[n] : 0.0f;
  const int mo = m0 + ((lane & 16) ? 8 : 0);
#pragma unroll
  for (int r = 0; r < 8; ++r) {
    float v = c[r] * scale + bv;
    if (relu) v = fmaxf(v, 0.0f);
    dst[(mo + r) * ld + n0 + n] = v;
  }
}

// ---------------- fused transformer kernel ----------------
// Grid: 512 blocks (one per batch item), 128 threads (4 wave32).
__global__ __launch_bounds__(128)
void transformer_fused(const float* __restrict__ x,
                       const float* __restrict__ w_in,  const float* __restrict__ b_in,
                       const float* __restrict__ w_pos, const float* __restrict__ b_pos,
                       const float* __restrict__ cls_token,
                       const float* __restrict__ attn_in_w,  const float* __restrict__ attn_in_b,
                       const float* __restrict__ attn_out_w, const float* __restrict__ attn_out_b,
                       const float* __restrict__ ln1_w, const float* __restrict__ ln1_b,
                       const float* __restrict__ ff1_w, const float* __restrict__ ff1_b,
                       const float* __restrict__ ff2_w, const float* __restrict__ ff2_b,
                       const float* __restrict__ ln2_w, const float* __restrict__ ln2_b,
                       const float* __restrict__ out_w, const float* __restrict__ out_b,
                       const float* __restrict__ mlp_w0, const float* __restrict__ mlp_b0,
                       const float* __restrict__ mlp_w1, const float* __restrict__ mlp_b1,
                       const float* __restrict__ mlp_w2, const float* __restrict__ mlp_b2,
                       float* __restrict__ out) {
  // LDS layout (floats): h[1536] | region1[6144] | region2[6144] | vec[576]
  __shared__ float smem[1536 + 6144 + 6144 + 576];
  float* sh_h   = smem;
  float* sh_r1  = smem + 1536;             // qkv(48x96) / proj(48x32) / FF sbuf (4x48x32)
  float* sh_r2  = smem + 1536 + 6144;      // scores(48x48)+attn(48x32) / FF partials (4x48x32)
  float* sh_vec = smem + 1536 + 6144 + 6144;

  const int bidx = blockIdx.x;
  const int tid  = threadIdx.x;
  const int lane = tid & 31;
  const int wave = tid >> 5;

  // ---- embed: h[s][d] = cls (s==0) else x*w_in + b_in + POS@w_pos^T + b_pos
  for (int e = tid; e < LLQ * DD; e += 128) {
    const int s = e >> 5, d = e & 31;
    float v;
    if (s == 0) {
      v = cls_token[d];
    } else {
      const int sp = s - 1;
      float pe = c_pos[sp * 3 + 0] * w_pos[d * 3 + 0] +
                 c_pos[sp * 3 + 1] * w_pos[d * 3 + 1] +
                 c_pos[sp * 3 + 2] * w_pos[d * 3 + 2] + b_pos[d];
      v = x[bidx * SS + sp] * w_in[d] + b_in[d] + pe;
    }
    sh_h[e] = v;
  }
  __syncthreads();

  // ---- 2 encoder layers ----
  for (int l = 0; l < 2; ++l) {
    const float* wi  = attn_in_w  + l * 96 * DD;
    const float* bi  = attn_in_b  + l * 96;
    const float* wo  = attn_out_w + l * DD * DD;
    const float* bo  = attn_out_b + l * DD;
    const float* l1w = ln1_w + l * DD;  const float* l1b = ln1_b + l * DD;
    const float* f1w = ff1_w + l * FFD * DD;  const float* f1b = ff1_b + l * FFD;
    const float* f2w = ff2_w + l * DD * FFD;  const float* f2b = ff2_b + l * DD;
    const float* l2w = ln2_w + l * DD;  const float* l2b = ln2_b + l * DD;

    // ---- QKV: r1[48][96] = h @ wi^T + bi  (3 M-tiles x 6 N-tiles, K=32)
    for (int t = wave; t < 18; t += 4) {
      const int mt = t / 6, nt = t % 6;
      v16h a = load_a16(sh_h, DD, mt * 16, 0, lane, 32);
      v16h b = load_bw16(wi, DD, nt * 16, 0, lane, 32, 16);
      v8f  c = {};
      c = wmma32(a, b, c);
      store_c(sh_r1, 96, mt * 16, nt * 16, lane, c, 16, 1.0f, bi + nt * 16, false);
    }
    __syncthreads();

    // ---- attention, heads processed sequentially
    float* sc   = sh_r2;          // 48x48 scores
    float* attn = sh_r2 + 2304;   // 48x32 concatenated head outputs
    for (int hh = 0; hh < HHN; ++hh) {
      // scores = (q_h @ k_h^T) / sqrt(8); only 8 K-slots valid, rest zero-padded
      for (int t = wave; t < 9; t += 4) {
        const int mt = t / 3, nt = t % 3;
        v16h a = load_a16(sh_r1 + hh * DH, 96, mt * 16, 0, lane, DH);
        v16h b = load_bw16(sh_r1 + 32 + hh * DH, 96, nt * 16, 0, lane, DH, 16);
        v8f  c = {};
        c = wmma32(a, b, c);
        store_c(sc, 48, mt * 16, nt * 16, lane, c, 16, 0.35355339059327379f, nullptr, false);
      }
      __syncthreads();
      // softmax over each row of 48
      if (tid < LLQ) {
        float* row = sc + tid * 48;
        float mx = row[0];
        for (int j = 1; j < 48; ++j) mx = fmaxf(mx, row[j]);
        float sum = 0.0f;
        for (int j = 0; j < 48; ++j) { float e2 = __expf(row[j] - mx); row[j] = e2; sum += e2; }
        const float inv = 1.0f / sum;
        for (int j = 0; j < 48; ++j) row[j] *= inv;
      }
      __syncthreads();
      // o_h = a(48x48) @ v_h(48x8): two K-chunks (32 + padded 16)
      if (wave < 3) {
        const int mt = wave;
        v8f c = {};
        v16h a0 = load_a16(sc, 48, mt * 16, 0, lane, 32);
        v16h b0 = load_bm16(sh_r1 + 64 + hh * DH, 96, 0, 0, lane, 32, DH);
        c = wmma32(a0, b0, c);
        v16h a1 = load_a16(sc, 48, mt * 16, 32, lane, 16);
        v16h b1 = load_bm16(sh_r1 + 64 + hh * DH, 96, 32, 0, lane, 16, DH);
        c = wmma32(a1, b1, c);
        store_c(attn, DD, mt * 16, hh * DH, lane, c, DH, 1.0f, nullptr, false);
      }
      __syncthreads();
    }

    // ---- out proj: r1[48][32] = attn @ wo^T + bo
    for (int t = wave; t < 6; t += 4) {
      const int mt = t / 2, nt = t % 2;
      v16h a = load_a16(attn, DD, mt * 16, 0, lane, 32);
      v16h b = load_bw16(wo, DD, nt * 16, 0, lane, 32, 16);
      v8f  c = {};
      c = wmma32(a, b, c);
      store_c(sh_r1, DD, mt * 16, nt * 16, lane, c, 16, 1.0f, bo + nt * 16, false);
    }
    __syncthreads();

    // ---- LN1: h = LN(h + proj)   (one row per thread)
    if (tid < LLQ) {
      float* hr = sh_h  + tid * DD;
      float* pr = sh_r1 + tid * DD;
      float xv[DD];
      float mu = 0.0f;
#pragma unroll
      for (int c = 0; c < DD; ++c) { xv[c] = hr[c] + pr[c]; mu += xv[c]; }
      mu *= (1.0f / DD);
      float var = 0.0f;
#pragma unroll
      for (int c = 0; c < DD; ++c) { float d = xv[c] - mu; var += d * d; }
      var *= (1.0f / DD);
      const float inv = rsqrtf(var + 1e-5f);
#pragma unroll
      for (int c = 0; c < DD; ++c) hr[c] = (xv[c] - mu) * inv * l1w[c] + l1b[c];
    }
    __syncthreads();

    // ---- FF: out += relu(h @ f1w^T + f1b) @ f2w^T, in 64 chunks of 32 FF
    //      channels; chunks partitioned across the 4 waves, partials in regs.
    //      Next chunk's weights are software-prefetched (global_prefetch_b8)
    //      to hide L2 latency at ~1 wave/SIMD occupancy.
    {
      float* sbuf = sh_r1 + wave * 1536;   // per-wave 48x32 relu staging
      float* part = sh_r2 + wave * 1536;   // per-wave 48x32 partial output
      v16h aH[3];
#pragma unroll
      for (int mt = 0; mt < 3; ++mt) aH[mt] = load_a16(sh_h, DD, mt * 16, 0, lane, 32);
      v8f acc[6] = {};
      // warm-up prefetch of this wave's first chunk
      __builtin_prefetch(f1w + (wave * 32 + (lane & 31)) * DD, 0, 1);
      __builtin_prefetch(f2w + (lane & 31) * FFD + wave * 32, 0, 1);
      for (int j = wave; j < 64; j += 4) {
        // prefetch next chunk's weight rows (each f1w row = 32 f32 = 128B line;
        // 32 lanes cover the 32 rows of each block)
        if (j + 4 < 64) {
          const int jn = j + 4;
          __builtin_prefetch(f1w + (jn * 32 + (lane & 31)) * DD, 0, 1);
          __builtin_prefetch(f2w + (lane & 31) * FFD + jn * 32, 0, 1);
        }
        // GEMM1: S = relu(h @ f1w_j^T + b)  -> sbuf (48x32)
#pragma unroll
        for (int t = 0; t < 6; ++t) {
          const int mt = t / 2, nt = t % 2;
          v16h b = load_bw16(f1w, DD, j * 32 + nt * 16, 0, lane, 32, 16);
          v8f  c = {};
          c = wmma32(aH[mt], b, c);
          store_c(sbuf, DD, mt * 16, nt * 16, lane, c, 16, 1.0f, f1b + j * 32 + nt * 16, true);
        }
        // GEMM2: acc += S @ W2_j   (W2_j[k][n] = f2w[n][j*32+k])
#pragma unroll
        for (int t = 0; t < 6; ++t) {
          const int mt = t / 2, nt = t % 2;
          v16h a = load_a16(sbuf, DD, mt * 16, 0, lane, 32);
          v16h b = load_bw16(f2w, FFD, nt * 16, j * 32, lane, 32, 16);
          acc[t] = wmma32(a, b, acc[t]);
        }
      }
#pragma unroll
      for (int t = 0; t < 6; ++t)
        store_c(part, DD, (t / 2) * 16, (t % 2) * 16, lane, acc[t], 16, 1.0f, nullptr, false);
    }
    __syncthreads();

    // ---- reduce partials + f2b + residual, then LN2 (one row per thread)
    if (tid < LLQ) {
      float* hr = sh_h + tid * DD;
      float xv[DD];
      float mu = 0.0f;
#pragma unroll
      for (int c = 0; c < DD; ++c) {
        float s = sh_r2[0 * 1536 + tid * DD + c] + sh_r2[1 * 1536 + tid * DD + c] +
                  sh_r2[2 * 1536 + tid * DD + c] + sh_r2[3 * 1536 + tid * DD + c];
        xv[c] = hr[c] + s + f2b[c];
        mu += xv[c];
      }
      mu *= (1.0f / DD);
      float var = 0.0f;
#pragma unroll
      for (int c = 0; c < DD; ++c) { float d = xv[c] - mu; var += d * d; }
      var *= (1.0f / DD);
      const float inv = rsqrtf(var + 1e-5f);
#pragma unroll
      for (int c = 0; c < DD; ++c) hr[c] = (xv[c] - mu) * inv * l2w[c] + l2b[c];
    }
    __syncthreads();
  }

  // ---- head: xvec[s] = relu(h[s+1] . out_w + out_b), s = 0..46
  if (tid < SS) {
    float s = out_b[0];
#pragma unroll
    for (int c = 0; c < DD; ++c) s += sh_h[(tid + 1) * DD + c] * out_w[c];
    sh_vec[tid] = fmaxf(s, 0.0f);
  }
  __syncthreads();

  // ---- MLP layer 0: 47 -> 256
  for (int jo = tid; jo < HIDD; jo += 128) {
    float s = mlp_b0[jo];
    for (int k = 0; k < SS; ++k) s += mlp_w0[jo * SS + k] * sh_vec[k];
    sh_vec[48 + jo] = fmaxf(s, 0.0f);
  }
  __syncthreads();

  // ---- MLP layer 1: 256 -> 256
  for (int jo = tid; jo < HIDD; jo += 128) {
    float s = mlp_b1[jo];
    for (int k = 0; k < HIDD; ++k) s += mlp_w1[jo * HIDD + k] * sh_vec[48 + k];
    sh_vec[304 + jo] = fmaxf(s, 0.0f);
  }
  __syncthreads();

  // ---- MLP layer 2: 256 -> 1 (with final ReLU)
  {
    float p = mlp_w2[tid] * sh_vec[304 + tid] + mlp_w2[tid + 128] * sh_vec[304 + tid + 128];
    sh_vec[tid] = p;  // xvec region is dead now
  }
  __syncthreads();
  if (tid == 0) {
    float s = mlp_b2[0];
    for (int k = 0; k < 128; ++k) s += sh_vec[k];
    out[bidx] = fmaxf(s, 0.0f);
  }
}

// ---------------- harness entry point ----------------
extern "C" void kernel_launch(void* const* d_in, const int* in_sizes, int n_in,
                              void* d_out, int out_size, void* d_ws, size_t ws_size,
                              hipStream_t stream) {
  (void)in_sizes; (void)n_in; (void)d_ws; (void)ws_size; (void)out_size;
  const float* x         = (const float*)d_in[0];
  const float* w_in      = (const float*)d_in[1];
  const float* b_in      = (const float*)d_in[2];
  const float* w_pos     = (const float*)d_in[3];
  const float* b_pos     = (const float*)d_in[4];
  const float* cls_token = (const float*)d_in[5];
  const float* attn_in_w  = (const float*)d_in[6];
  const float* attn_in_b  = (const float*)d_in[7];
  const float* attn_out_w = (const float*)d_in[8];
  const float* attn_out_b = (const float*)d_in[9];
  const float* ln1_w = (const float*)d_in[10];
  const float* ln1_b = (const float*)d_in[11];
  const float* ff1_w = (const float*)d_in[12];
  const float* ff1_b = (const float*)d_in[13];
  const float* ff2_w = (const float*)d_in[14];
  const float* ff2_b = (const float*)d_in[15];
  const float* ln2_w = (const float*)d_in[16];
  const float* ln2_b = (const float*)d_in[17];
  const float* out_w = (const float*)d_in[18];
  const float* out_b = (const float*)d_in[19];
  const float* mlp_w0 = (const float*)d_in[20];
  const float* mlp_b0 = (const float*)d_in[21];
  const float* mlp_w1 = (const float*)d_in[22];
  const float* mlp_b1 = (const float*)d_in[23];
  const float* mlp_w2 = (const float*)d_in[24];
  const float* mlp_b2 = (const float*)d_in[25];

  transformer_fused<<<dim3(BB), dim3(128), 0, stream>>>(
      x, w_in, b_in, w_pos, b_pos, cls_token,
      attn_in_w, attn_in_b, attn_out_w, attn_out_b,
      ln1_w, ln1_b, ff1_w, ff1_b, ff2_w, ff2_b, ln2_w, ln2_b,
      out_w, out_b, mlp_w0, mlp_b0, mlp_w1, mlp_b1, mlp_w2, mlp_b2,
      (float*)d_out);
}
